// MRFCodebookGenerator_18794776887805
// MI455X (gfx1250) — matrix operand order repaired and checked by reading, not compiled, not run on hardware
//
#include <hip/hip_runtime.h>
#include <hip/hip_bf16.h>
#include <math.h>

// MI455X / gfx1250 persistent single-WGP kernel for the MRF codebook scan.
// Serial 256-step chain -> latency-bound: one workgroup, 32 wave32 waves,
// all recurrent state in LDS, two-body term as 128x1280x32 GEMM via
// v_wmma_f32_16x16x32_f16 (inner dim L*S=32 == WMMA K), Gumbel-max sampling.

typedef __attribute__((ext_vector_type(16))) _Float16 v16h;
typedef __attribute__((ext_vector_type(8)))  float    v8f;

#define LBITS    16
#define T_TYPES  256
#define BATCH    128
#define WWIN     32          // two-body window == L*S == WMMA K
#define PTOP     1024
#define RRAND    256
#define KPROP    (PTOP + RRAND)   // 1280 proposals per column
#define NTHREADS 1024
#define NWAVES   32

__device__ __forceinline__ unsigned hash_u32(unsigned x) {
  x ^= x >> 16; x *= 0x7feb352dU;
  x ^= x >> 15; x *= 0x846ca68bU;
  x ^= x >> 16; return x;
}

__global__ __launch_bounds__(NTHREADS)
void mrf_codebook_gfx1250(const float* __restrict__ pi_t,
                          float* __restrict__ out_c,     // [128,256,16,2]
                          float* __restrict__ out_lp,    // [128]
                          float* __restrict__ ws_energy) // [128,1280] f32 scratch
{
  __shared__ float          pi_lds[T_TYPES];
  __shared__ float          vstate[BATCH][WWIN];     // v[b,ls] = sum_w pi_w * c[b,w,ls]
  __shared__ unsigned short hist[BATCH][WWIN];       // ring buffer of sampled codes
  __shared__ unsigned short prop_idx[KPROP];
  __shared__ float          ob_k[KPROP];
  __shared__ int            binom[17][17];
  __shared__ int            cls_order[17];
  __shared__ int            cls_cum[17];
  __shared__ float          cls_e[17];
  __shared__ unsigned short sel_idx[BATCH];
  __shared__ unsigned short old_idx[BATCH];
  __shared__ float          lp_acc[BATCH];

  const int tid  = threadIdx.x;
  const int lane = tid & 31;
  const int wave = tid >> 5;

  // ---- init ----
  if (tid < T_TYPES) pi_lds[tid] = pi_t[tid];
  for (int p = tid; p < BATCH * WWIN; p += NTHREADS) {
    (&vstate[0][0])[p] = 0.0f;
    (&hist[0][0])[p]   = 0;
  }
  if (tid < BATCH) lp_acc[tid] = 0.0f;
  if (tid == 0) {
    for (int n = 0; n <= 16; ++n) {
      binom[n][0] = 1;
      for (int k = 1; k <= 16; ++k)
        binom[n][k] = (n == 0) ? 0 : (binom[n-1][k-1] + binom[n-1][k]);
    }
  }
  __builtin_prefetch(pi_t, 0, 3);          // global_prefetch_b8
  __syncthreads();

  const float inv_l = 1.0f / (float)LBITS;

  for (int i = 0; i < T_TYPES; ++i) {
    const float pi_i   = pi_lds[i];
    const float target = (float)LBITS * (1.0f - (float)i / (float)T_TYPES);

    // ---- Phase A: exact top-P via popcount-class sort + combination unranking
    if (tid == 0) {
      float e[17]; int ord[17];
      for (int w = 0; w <= 16; ++w) {
        float d = (float)w - target;
        e[w] = pi_i * d * d * inv_l;
        ord[w] = w; cls_e[w] = e[w];
      }
      for (int a = 1; a <= 16; ++a) {       // insertion sort, energy ascending
        int ov = ord[a]; float ev = e[ov]; int b2 = a - 1;
        while (b2 >= 0 && e[ord[b2]] > ev) { ord[b2+1] = ord[b2]; --b2; }
        ord[b2+1] = ov;
      }
      int cum = 0;
      for (int j = 0; j <= 16; ++j) {
        cls_order[j] = ord[j];
        cum += binom[16][ord[j]];
        cls_cum[j] = cum;
      }
    }
    __syncthreads();

    // ---- Phase B: fill 1280 proposals (top by rank-unranking, plus random)
    for (int kk = tid; kk < KPROP; kk += NTHREADS) {
      int idxv; float ob;
      if (kk < PTOP) {
        int r = kk, j = 0;
        while (cls_cum[j] <= r) ++j;
        int pos = r - (j ? cls_cum[j-1] : 0);
        int w   = cls_order[j];
        int id = 0, rem = w;                 // unrank pos-th value of popcount w
        for (int bit = 15; bit >= 0 && rem > 0; --bit) {
          int c = binom[bit][rem];
          if (pos >= c) { pos -= c; id |= (1 << bit); --rem; }
        }
        idxv = id;
        ob   = cls_e[w];
      } else {
        unsigned h = hash_u32(0x9E3779B9u ^ (unsigned)(i * 7919 + kk));
        idxv = (int)(h & 0xFFFFu);
        int w = __popc((unsigned)idxv);
        float d = (float)w - target;
        ob = pi_i * d * d * inv_l;
      }
      prop_idx[kk] = (unsigned short)idxv;
      ob_k[kk]     = ob;
    }
    __syncthreads();

    // ---- Phase C: two-body GEMM  e_bk = -(ob_k + scale * (v_b . code_k))
    // 8 m-tiles x 80 n-tiles = 640 WMMA tiles = 20 per wave.
    float nint   = fminf((float)i, (float)WWIN);
    float factor = (i > 0) ? ((float)i / fmaxf(nint, 1.0f)) : 0.0f;
    const float scale = factor * pi_i;

    const int  mt   = wave >> 2;
    const int  ntb  = (wave & 3) * 20;
    const int  mrow = lane & 15;
    const bool hiL  = lane >= 16;

    union { v16h v; _Float16 h[16]; } afrag;   // A 16x32 f16: lanes0-15 K=0..15, lanes16-31 K=16..31
    {
      int b = mt * 16 + mrow;
      #pragma unroll
      for (int j = 0; j < 8; ++j) {
        int k0 = (hiL ? 16 : 0) + 2 * j;
        afrag.h[2*j]   = (_Float16)vstate[b][k0];
        afrag.h[2*j+1] = (_Float16)vstate[b][k0 + 1];
      }
    }
    // Unroll 4: gives the scheduler independent fragment builds to co-execute
    // in the WMMA->VALU hazard slots (avoids the 4 v_nop bubbles).
    #pragma unroll 4
    for (int nt = ntb; nt < ntb + 20; ++nt) {
      int prop = nt * 16 + mrow;
      int cidx = prop_idx[prop];
      // B 32x16 f16: lanes0-15 hold K=0..15, lanes16-31 K=16..31; VGPR j packs
      // K={2j,2j+1} = one-hot pair of code bit l=(hi?8:0)+j -> complementary
      // halves, so each DWORD is a single select between two f16-pair constants.
      union { v16h v; unsigned u[8]; } bfrag;
      #pragma unroll
      for (int j = 0; j < 8; ++j) {
        int l = (hiL ? 8 : 0) + j;
        bfrag.u[j] = ((cidx >> (15 - l)) & 1) ? 0x3C000000u : 0x00003C00u;
      }
      v8f acc = {};
      acc = __builtin_amdgcn_wmma_f32_16x16x32_f16(
          false, afrag.v, false, bfrag.v, (short)0, acc, false, false);
      float obv = ob_k[prop];
      #pragma unroll
      for (int j = 0; j < 8; ++j) {
        int b = mt * 16 + j + (hiL ? 8 : 0);           // C/D layout: M=j (+8 hi)
        ws_energy[b * KPROP + prop] = -(obv + scale * acc[j]);
      }
    }
    __threadfence();
    __syncthreads();

    // ---- Phase D: Gumbel-max sampling + logsumexp, 4 batches per wave
    for (int bb = 0; bb < 4; ++bb) {
      int b = wave * 4 + bb;
      const float* erow = ws_energy + (size_t)b * KPROP;
      float vmax = -1e30f, bs = -1e30f, be = 0.0f; int bk = 0;
      for (int k = lane; k < KPROP; k += 32) {
        float e = erow[k];
        vmax = fmaxf(vmax, e);
        unsigned h = hash_u32(
            (unsigned)((i * BATCH + b) * KPROP + k) * 2654435761u + 0x68bc21ebu);
        float u = ((float)(h >> 8) + 0.5f) * (1.0f / 16777216.0f);
        float g = -logf(-logf(u));
        float sc = e + g;
        if (sc > bs) { bs = sc; be = e; bk = k; }
      }
      #pragma unroll
      for (int off = 16; off > 0; off >>= 1) {
        float os = __shfl_xor(bs, off, 32);
        float oe = __shfl_xor(be, off, 32);
        int   ok = __shfl_xor(bk, off, 32);
        float om = __shfl_xor(vmax, off, 32);
        vmax = fmaxf(vmax, om);
        if (os > bs || (os == bs && ok < bk)) { bs = os; be = oe; bk = ok; }
      }
      float sum = 0.0f;
      for (int k = lane; k < KPROP; k += 32) sum += __expf(erow[k] - vmax);
      #pragma unroll
      for (int off = 16; off > 0; off >>= 1) sum += __shfl_xor(sum, off, 32);
      if (lane == 0) {
        unsigned short sidx = prop_idx[bk];
        int slot = i & (WWIN - 1);
        old_idx[b]    = hist[b][slot];
        hist[b][slot] = sidx;
        sel_idx[b]    = sidx;
        lp_acc[b]    += be - (vmax + logf(sum));
      }
    }
    __syncthreads();

    // ---- Phase E: sliding-window state update + emit output column
    for (int p = tid; p < BATCH * 32; p += NTHREADS) {
      int b = p >> 5, ls = p & 31;
      int l = ls >> 1, s = ls & 1;
      int ni = sel_idx[b];
      float cv = (((ni >> (15 - l)) & 1) == s) ? 1.0f : 0.0f;
      float v = vstate[b][ls] + pi_i * cv;
      if (i >= WWIN) {
        int oi = old_idx[b];
        v -= pi_lds[i - WWIN] * ((((oi >> (15 - l)) & 1) == s) ? 1.0f : 0.0f);
      }
      vstate[b][ls] = v;
      out_c[((size_t)b * T_TYPES + i) * 32 + ls] = cv;
    }
    __syncthreads();
  }

  if (tid < BATCH) out_lp[tid] = lp_acc[tid] * (1.0f / (float)T_TYPES);
}

extern "C" void kernel_launch(void* const* d_in, const int* in_sizes, int n_in,
                              void* d_out, int out_size, void* d_ws, size_t ws_size,
                              hipStream_t stream) {
  (void)in_sizes; (void)n_in; (void)out_size; (void)ws_size;
  const float* pi = (const float*)d_in[0];
  float* out    = (float*)d_out;
  float* out_c  = out;                                   // [128,256,16,2]
  float* out_lp = out + (size_t)BATCH * T_TYPES * 32;    // [128]
  float* ws     = (float*)d_ws;                          // >= 128*1280*4 B
  hipLaunchKernelGGL(mrf_codebook_gfx1250, dim3(1), dim3(NTHREADS), 0, stream,
                     pi, out_c, out_lp, ws);
}
